// LongformerSegmentClassifier_64527588655341
// MI455X (gfx1250) — matrix operand order repaired and checked by reading, not compiled, not run on hardware
//
#include <hip/hip_runtime.h>
#include <hip/hip_bf16.h>
#include <cstdint>

// ---------------- problem constants ----------------
#define Bc   2
#define Sc   4096
#define DMc  768
#define DFFc 3072
#define Hc   12
#define Dc   64
#define Wc   256
#define Lc   2
#define NBc  16              // Sc / Wc
#define TWc  768             // 3*Wc
#define SPc  (Sc + 2 * Wc)   // padded seq for vT
#define NEGV (-1.0e9f)

typedef __bf16 bf16;
typedef __attribute__((ext_vector_type(16))) __bf16 v16bf;
typedef __attribute__((ext_vector_type(8)))  float  v8f;

// gfx1250 async global->LDS staging (ASYNCcnt path). Guarded so the host pass
// and older toolchains fall back to global_load_b128 + ds_store_b128.
#if __has_builtin(__builtin_amdgcn_global_load_async_to_lds_b128)
#define HAVE_ASYNC_LDS 1
#else
#define HAVE_ASYNC_LDS 0
#endif

#if HAVE_ASYNC_LDS
// builtin signature (from clang diagnostic): pointers to int __vector_size__(16)
typedef int v4i_vec __attribute__((vector_size(4 * sizeof(int))));
typedef __attribute__((address_space(1))) v4i_vec* glb_v4i;
typedef __attribute__((address_space(3))) v4i_vec* lds_v4i;
__device__ __forceinline__ glb_v4i as_global(const void* p) {
  return (glb_v4i)(void*)p;
}
__device__ __forceinline__ lds_v4i as_lds(void* p) {
  return (lds_v4i)p;
}
__device__ __forceinline__ void async_wait0() {
#if __has_builtin(__builtin_amdgcn_s_wait_asynccnt)
  __builtin_amdgcn_s_wait_asynccnt(0);
#else
  asm volatile("s_wait_asynccnt 0" ::: "memory");
#endif
}
#endif

union U4BF8 { uint4 u; bf16 h[8]; };

// Build a 16-element bf16 WMMA operand fragment from two 16B-aligned chunks.
__device__ __forceinline__ v16bf mk_frag(const bf16* p0, const bf16* p1) {
  U4BF8 a, b;
  a.u = *(const uint4*)p0;
  b.u = *(const uint4*)p1;
  v16bf f;
#pragma unroll
  for (int i = 0; i < 8; i++) { f[i] = a.h[i]; f[8 + i] = b.h[i]; }
  return f;
}

__device__ __forceinline__ v16bf zero_frag() {
  v16bf f;
#pragma unroll
  for (int i = 0; i < 16; i++) f[i] = (bf16)0.0f;
  return f;
}

__device__ __forceinline__ v8f zero_acc() {
  v8f a;
#pragma unroll
  for (int i = 0; i < 8; i++) a[i] = 0.0f;
  return a;
}

__device__ __forceinline__ v8f wmma_bf16(v16bf a, v16bf b, v8f c) {
  // D = A(16x32) * B(32x16) + C, f32 accumulate
  return __builtin_amdgcn_wmma_f32_16x16x32_bf16(false, a, false, b, (short)0, c,
                                                 false, false);
}

__device__ __forceinline__ float gelu_tanh(float x) {
  float x3 = x * x * x;
  return 0.5f * x * (1.0f + tanhf(0.7978845608028654f * (x + 0.044715f * x3)));
}

__device__ __forceinline__ float block_sum256(float v, float* rbuf) {
  int t = threadIdx.x;
  rbuf[t] = v;
  __syncthreads();
  for (int off = 128; off > 0; off >>= 1) {
    if (t < off) rbuf[t] += rbuf[t + off];
    __syncthreads();
  }
  float r = rbuf[0];
  __syncthreads();
  return r;
}

// ---------------- embedding + LN ----------------
__global__ __launch_bounds__(256) void k_embed_ln(const int* __restrict__ ids,
                                                  const float* __restrict__ wemb,
                                                  const float* __restrict__ pemb,
                                                  const float* __restrict__ gs,
                                                  const float* __restrict__ gb,
                                                  float* __restrict__ xf,
                                                  bf16* __restrict__ xh) {
  __shared__ float rbuf[256];
  int r = blockIdx.x;  // b*S + s
  int s = r % Sc;
  int t = threadIdx.x;
  int id = ids[r];
  float v[3];
#pragma unroll
  for (int i = 0; i < 3; i++) {
    int c = t + i * 256;
    v[i] = wemb[(size_t)id * DMc + c] + pemb[(size_t)s * DMc + c];
  }
  float mean = block_sum256(v[0] + v[1] + v[2], rbuf) * (1.0f / DMc);
  float sq = 0.0f;
#pragma unroll
  for (int i = 0; i < 3; i++) { float d = v[i] - mean; sq += d * d; }
  float var = block_sum256(sq, rbuf) * (1.0f / DMc);
  float rstd = rsqrtf(var + 1e-5f);
#pragma unroll
  for (int i = 0; i < 3; i++) {
    int c = t + i * 256;
    float y = (v[i] - mean) * rstd * gs[c] + gb[c];
    xf[(size_t)r * DMc + c] = y;
    xh[(size_t)r * DMc + c] = (bf16)y;
  }
}

// ---------------- residual add + LN ----------------
__global__ __launch_bounds__(256) void k_add_ln(const float* __restrict__ res,
                                                const float* __restrict__ add,
                                                const float* __restrict__ gs,
                                                const float* __restrict__ gb,
                                                float* __restrict__ xf,
                                                bf16* __restrict__ xh) {
  __shared__ float rbuf[256];
  int r = blockIdx.x;
  int t = threadIdx.x;
  float v[3];
#pragma unroll
  for (int i = 0; i < 3; i++) {
    int c = t + i * 256;
    v[i] = res[(size_t)r * DMc + c] + add[(size_t)r * DMc + c];
  }
  float mean = block_sum256(v[0] + v[1] + v[2], rbuf) * (1.0f / DMc);
  float sq = 0.0f;
#pragma unroll
  for (int i = 0; i < 3; i++) { float d = v[i] - mean; sq += d * d; }
  float var = block_sum256(sq, rbuf) * (1.0f / DMc);
  float rstd = rsqrtf(var + 1e-5f);
#pragma unroll
  for (int i = 0; i < 3; i++) {
    int c = t + i * 256;
    float y = (v[i] - mean) * rstd * gs[c] + gb[c];
    xf[(size_t)r * DMc + c] = y;
    xh[(size_t)r * DMc + c] = (bf16)y;
  }
}

// ---------------- weight transpose f32[K,N] -> bf16[N,K] ----------------
__global__ __launch_bounds__(256) void k_transpose_w(const float* __restrict__ in,
                                                     bf16* __restrict__ out, int K,
                                                     int N) {
  int idx = blockIdx.x * 256 + threadIdx.x;
  if (idx >= K * N) return;
  int n = idx / K, k = idx % K;
  out[idx] = (bf16)in[(size_t)k * N + n];
}

// ---------------- generic GEMM: C = act(alpha*(A[MxK] x Bt[NxK]^T + bias)) ----
// 256 threads = 8 waves; block tile 128x128; wave tile 64x32; K-step 32.
__global__ __launch_bounds__(256) void k_gemm_bt(const bf16* __restrict__ A,
                                                 const bf16* __restrict__ Bt,
                                                 const float* __restrict__ bias,
                                                 float* outF, bf16* outH, int M,
                                                 int N, int K, float alpha,
                                                 int gelu) {
  __shared__ alignas(16) bf16 As[128][48];  // 96B row stride: b128-aligned, low-conflict
  __shared__ alignas(16) bf16 Bs[128][48];
  int t = threadIdx.x;
  int m0 = blockIdx.x * 128, n0 = blockIdx.y * 128;
  int wave = t >> 5, lane = t & 31, half = lane >> 4, l16 = lane & 15;
  int wm = wave >> 2, wn = wave & 3;

  v8f acc[4][2];
#pragma unroll
  for (int i = 0; i < 4; i++)
#pragma unroll
    for (int j = 0; j < 2; j++) acc[i][j] = zero_acc();

  for (int kb = 0; kb < K; kb += 32) {
    // cooperative stage: 128x32 bf16 tiles, 16B chunks
    for (int c = t; c < 512; c += 256) {
      int row = c >> 2, kc = (c & 3) << 3;
      int gm = m0 + row;
      int gn = n0 + row;
#if HAVE_ASYNC_LDS
      if (gm < M) {
        __builtin_amdgcn_global_load_async_to_lds_b128(
            as_global(A + (size_t)gm * K + kb + kc), as_lds(&As[row][kc]), 0, 0);
      } else {
        *(uint4*)&As[row][kc] = make_uint4(0u, 0u, 0u, 0u);
      }
      __builtin_amdgcn_global_load_async_to_lds_b128(
          as_global(Bt + (size_t)gn * K + kb + kc), as_lds(&Bs[row][kc]), 0, 0);
#else
      uint4 av = make_uint4(0u, 0u, 0u, 0u);
      if (gm < M) av = *(const uint4*)(A + (size_t)gm * K + kb + kc);
      *(uint4*)&As[row][kc] = av;
      *(uint4*)&Bs[row][kc] = *(const uint4*)(Bt + (size_t)gn * K + kb + kc);
#endif
    }
    if (kb + 32 < K) {  // gfx1250 global_prefetch of next K slab
      __builtin_prefetch(A + (size_t)(m0 + (t >> 1)) * K + kb + 32, 0, 1);
      __builtin_prefetch(Bt + (size_t)(n0 + (t >> 1)) * K + kb + 32, 0, 1);
    }
#if HAVE_ASYNC_LDS
    async_wait0();  // drain ASYNCcnt before the workgroup barrier
#endif
    __syncthreads();

    v16bf bfr[2];
#pragma unroll
    for (int nt = 0; nt < 2; nt++) {
      const bf16* bp = &Bs[wn * 32 + nt * 16 + l16][0];
      bfr[nt] = mk_frag(bp + half * 8, bp + 16 + half * 8);
    }
#pragma unroll
    for (int mt = 0; mt < 4; mt++) {
      const bf16* ap = &As[wm * 64 + mt * 16 + l16][0];
      v16bf af = mk_frag(ap + half * 8, ap + 16 + half * 8);
#pragma unroll
      for (int nt = 0; nt < 2; nt++) acc[mt][nt] = wmma_bf16(af, bfr[nt], acc[mt][nt]);
    }
    __syncthreads();
  }

#pragma unroll
  for (int mt = 0; mt < 4; mt++)
#pragma unroll
    for (int nt = 0; nt < 2; nt++)
#pragma unroll
      for (int r = 0; r < 8; r++) {
        int gr = m0 + wm * 64 + mt * 16 + r + 8 * half;
        int gc = n0 + wn * 32 + nt * 16 + l16;
        if (gr < M) {
          float v = (acc[mt][nt][r] + bias[gc]) * alpha;
          if (gelu) v = gelu_tanh(v);
          if (outF) outF[(size_t)gr * N + gc] = v;
          if (outH) outH[(size_t)gr * N + gc] = (bf16)v;
        }
      }
}

// ---------------- v -> vT[b,h,d, W+kpos] (zero padded window) ----------------
__global__ __launch_bounds__(256) void k_build_vT(const bf16* __restrict__ vh,
                                                  bf16* __restrict__ vT) {
  int idx = blockIdx.x * 256 + threadIdx.x;
  const int total = Bc * Hc * Dc * SPc;
  if (idx >= total) return;
  int sp = idx % SPc;
  int d = (idx / SPc) % Dc;
  int h = (idx / (SPc * Dc)) % Hc;
  int b = idx / (SPc * Dc * Hc);
  int kpos = sp - Wc;
  bf16 v = (bf16)0.0f;
  if (kpos >= 0 && kpos < Sc) v = vh[((size_t)b * Sc + kpos) * DMc + h * Dc + d];
  vT[idx] = v;
}

// ---------------- banded local attention + global-token column ----------------
// one block per (b, h, nb); 256 threads = 8 waves
__global__ __launch_bounds__(256) void k_band_attn(const bf16* __restrict__ qh,
                                                   const bf16* __restrict__ kh,
                                                   const bf16* __restrict__ vh,
                                                   const bf16* __restrict__ vT,
                                                   const int* __restrict__ am,
                                                   bf16* __restrict__ oh) {
  __shared__ float sc[16][776];   // 769 used (3W band + global col)
  __shared__ float osh[16][64];   // cross-wave output reduction
  __shared__ float red[16][16];
  __shared__ float pg[16];
  int blk = blockIdx.x;
  int nb = blk % NBc;
  int h = (blk / NBc) % Hc;
  int b = blk / (NBc * Hc);
  int t = threadIdx.x;
  int wave = t >> 5, lane = t & 31, half = lane >> 4, l16 = lane & 15;
  const int am0 = am[(size_t)b * Sc] > 0;

  for (int mt = 0; mt < 16; mt++) {  // 16 M-tiles cover the W=256 query rows
    // -------- phase 1: scores = q . k^T (WMMA) + global column (VALU) --------
    if (t < 16) {
      int row = t;
      int s = nb * Wc + mt * 16 + row;
      const bf16* qr = qh + ((size_t)b * Sc + s) * DMc + h * Dc;
      const bf16* k0p = kh + (size_t)b * Sc * DMc + h * Dc;
      float acc = 0.0f;
#pragma unroll
      for (int d = 0; d < Dc; d++) acc += (float)qr[d] * (float)k0p[d];
      sc[row][TWc] = am0 ? acc : NEGV;
    }
    for (int nt = wave; nt < 48; nt += 8) {
      v8f acc = zero_acc();
      int kj = nt * 16 + l16;
      int kpos = nb * Wc - Wc + kj;
#pragma unroll
      for (int ks = 0; ks < 2; ks++) {
        int k0 = ks * 32;
        int s = nb * Wc + mt * 16 + l16;
        const bf16* ap = qh + ((size_t)b * Sc + s) * DMc + h * Dc + k0;
        v16bf af = mk_frag(ap + half * 8, ap + 16 + half * 8);
        v16bf bfb;
        if (kpos >= 0 && kpos < Sc) {
          const bf16* bp = kh + ((size_t)b * Sc + kpos) * DMc + h * Dc + k0;
          bfb = mk_frag(bp + half * 8, bp + 16 + half * 8);
        } else {
          bfb = zero_frag();
        }
        acc = wmma_bf16(af, bfb, acc);
      }
#pragma unroll
      for (int r = 0; r < 8; r++) {
        int row = r + 8 * half;
        int qi = mt * 16 + row;
        bool ok = (kj >= qi) && (kj <= qi + 2 * Wc) && (kpos >= 0) && (kpos < Sc) &&
                  (kpos != 0);
        if (ok) ok = am[(size_t)b * Sc + kpos] > 0;
        sc[row][kj] = ok ? acc[r] : NEGV;
      }
    }
    __syncthreads();

    // -------- phase 2: row softmax over 769 entries --------
    {
      int row = t >> 4, j0 = t & 15;
      float m = -3.0e38f;
      for (int j = j0; j <= TWc; j += 16) m = fmaxf(m, sc[row][j]);
      red[row][j0] = m;
      __syncthreads();
      float rm = -3.0e38f;
#pragma unroll
      for (int i = 0; i < 16; i++) rm = fmaxf(rm, red[row][i]);
      __syncthreads();
      float se = 0.0f;
      for (int j = j0; j <= TWc; j += 16) {
        float e = __expf(sc[row][j] - rm);
        sc[row][j] = e;
        se += e;
      }
      red[row][j0] = se;
      __syncthreads();
      float rs = 0.0f;
#pragma unroll
      for (int i = 0; i < 16; i++) rs += red[row][i];
      float inv = 1.0f / rs;
      for (int j = j0; j <= TWc; j += 16) sc[row][j] *= inv;
      if (j0 == 0) pg[row] = sc[row][TWc];
    }
    __syncthreads();
    for (int i = t; i < 1024; i += 256) ((float*)osh)[i] = 0.0f;
    __syncthreads();

    // -------- phase 3: out = P x Vband (WMMA, wave-split over K) --------
    {
      v8f acc[4];
#pragma unroll
      for (int nt = 0; nt < 4; nt++) acc[nt] = zero_acc();
#pragma unroll
      for (int kk = 0; kk < 3; kk++) {
        int kbs = wave * 96 + kk * 32;
        v16bf af;
#pragma unroll
        for (int i = 0; i < 8; i++) {
          af[i] = (bf16)sc[l16][kbs + half * 8 + i];
          af[8 + i] = (bf16)sc[l16][kbs + 16 + half * 8 + i];
        }
#pragma unroll
        for (int nt = 0; nt < 4; nt++) {
          int d = nt * 16 + l16;
          const bf16* bp =
              vT + (((size_t)(b * Hc + h) * Dc + d) * SPc + nb * Wc + kbs);
          v16bf bfb = mk_frag(bp + half * 8, bp + 16 + half * 8);
          acc[nt] = wmma_bf16(af, bfb, acc[nt]);
        }
      }
#pragma unroll
      for (int nt = 0; nt < 4; nt++)
#pragma unroll
        for (int r = 0; r < 8; r++)
          atomicAdd(&osh[r + 8 * half][nt * 16 + l16], acc[nt][r]);
    }
    __syncthreads();

    // -------- phase 4: add global-token value, store bf16 --------
    {
      int row = t >> 4, c0 = (t & 15) * 4;
      int s = nb * Wc + mt * 16 + row;
      const bf16* v0 = vh + (size_t)b * Sc * DMc + h * Dc;
#pragma unroll
      for (int cc = 0; cc < 4; cc++) {
        int d = c0 + cc;
        float val = osh[row][d] + pg[row] * (float)v0[d];
        oh[((size_t)b * Sc + s) * DMc + h * Dc + d] = (bf16)val;
      }
    }
    __syncthreads();
  }
}

// ---------------- global attention for token 0 (overwrites row s=0) ----------
__global__ __launch_bounds__(256) void k_global_attn(const float* __restrict__ qg,
                                                     const bf16* __restrict__ kgh,
                                                     const bf16* __restrict__ vgh,
                                                     const int* __restrict__ am,
                                                     bf16* __restrict__ oh) {
  __shared__ float sgl[Sc];
  __shared__ float red[256];
  __shared__ float part[4][64];
  int blk = blockIdx.x;
  int h = blk % Hc, b = blk / Hc;
  int t = threadIdx.x;
  float qv[64];
#pragma unroll
  for (int d = 0; d < 64; d++) qv[d] = qg[(size_t)b * DMc + h * Dc + d];
  float lm = -3.0e38f;
  for (int s = t; s < Sc; s += 256) {
    const bf16* kr = kgh + ((size_t)b * Sc + s) * DMc + h * Dc;
    float a = 0.0f;
#pragma unroll
    for (int d = 0; d < 64; d++) a += qv[d] * (float)kr[d];
    float v = (am[(size_t)b * Sc + s] > 0) ? a : NEGV;
    sgl[s] = v;
    lm = fmaxf(lm, v);
  }
  red[t] = lm;
  __syncthreads();
  for (int off = 128; off > 0; off >>= 1) {
    if (t < off) red[t] = fmaxf(red[t], red[t + off]);
    __syncthreads();
  }
  float m = red[0];
  __syncthreads();
  float ls = 0.0f;
  for (int s = t; s < Sc; s += 256) {
    float e = __expf(sgl[s] - m);
    sgl[s] = e;
    ls += e;
  }
  red[t] = ls;
  __syncthreads();
  for (int off = 128; off > 0; off >>= 1) {
    if (t < off) red[t] += red[t + off];
    __syncthreads();
  }
  float inv = 1.0f / red[0];
  int g = t >> 6, d = t & 63;
  float p = 0.0f;
  for (int s = g; s < Sc; s += 4)
    p += sgl[s] * (float)vgh[((size_t)b * Sc + s) * DMc + h * Dc + d];
  part[g][d] = p * inv;
  __syncthreads();
  if (t < 64) {
    float o = part[0][t] + part[1][t] + part[2][t] + part[3][t];
    oh[(size_t)b * Sc * DMc + h * Dc + t] = (bf16)o;
  }
}

// ---------------- pad x[:,0,:] into a 16-row bf16 matrix for qg GEMM ---------
__global__ __launch_bounds__(256) void k_x0pad(const bf16* __restrict__ xh,
                                               bf16* __restrict__ x0) {
  int idx = blockIdx.x * 256 + threadIdx.x;
  if (idx >= 16 * DMc) return;
  int row = idx / DMc, c = idx % DMc;
  x0[idx] = (row < Bc) ? xh[(size_t)row * Sc * DMc + c] : (bf16)0.0f;
}

// ---------------- classifier: out[b,c] = x[b,0,:] . Wcls[:,c] + bcls ---------
__global__ __launch_bounds__(256) void k_cls(const float* __restrict__ xf,
                                             const float* __restrict__ Wcls,
                                             const float* __restrict__ bcls,
                                             float* __restrict__ out) {
  __shared__ float p[256];
  int t = threadIdx.x;
  int b = t >> 7, c = (t >> 6) & 1, k0 = t & 63;
  float s = 0.0f;
  for (int k = k0; k < DMc; k += 64)
    s += xf[(size_t)b * Sc * DMc + k] * Wcls[(size_t)k * 2 + c];
  p[t] = s;
  __syncthreads();
  if ((t & 63) == 0) {
    float tot = 0.0f;
    for (int i = 0; i < 64; i++) tot += p[t + i];
    out[b * 2 + c] = tot + bcls[c];
  }
}

// ======================= host-side orchestration =============================
extern "C" void kernel_launch(void* const* d_in, const int* in_sizes, int n_in,
                              void* d_out, int out_size, void* d_ws, size_t ws_size,
                              hipStream_t stream) {
  (void)in_sizes; (void)n_in; (void)out_size; (void)ws_size;
  const int*   ids   = (const int*)d_in[0];
  const int*   am    = (const int*)d_in[1];
  const float* wemb  = (const float*)d_in[2];
  const float* pemb  = (const float*)d_in[3];
  const float* ln_es = (const float*)d_in[4];
  const float* ln_eb = (const float*)d_in[5];
  const float* Wq  = (const float*)d_in[6];   const float* bq  = (const float*)d_in[7];
  const float* Wk  = (const float*)d_in[8];   const float* bk  = (const float*)d_in[9];
  const float* Wv  = (const float*)d_in[10];  const float* bv  = (const float*)d_in[11];
  const float* Wqg = (const float*)d_in[12];  const float* bqg = (const float*)d_in[13];
  const float* Wkg = (const float*)d_in[14];  const float* bkg = (const float*)d_in[15];
  const float* Wvg = (const float*)d_in[16];  const float* bvg = (const float*)d_in[17];
  const float* Wo  = (const float*)d_in[18];  const float* bo  = (const float*)d_in[19];
  const float* l1s = (const float*)d_in[20];  const float* l1b = (const float*)d_in[21];
  const float* W1  = (const float*)d_in[22];  const float* b1  = (const float*)d_in[23];
  const float* W2  = (const float*)d_in[24];  const float* b2  = (const float*)d_in[25];
  const float* l2s = (const float*)d_in[26];  const float* l2b = (const float*)d_in[27];
  const float* Wcls = (const float*)d_in[28]; const float* bcls = (const float*)d_in[29];
  float* out = (float*)d_out;

  // workspace carve-up (~208 MB)
  char* ws = (char*)d_ws;
  size_t off = 0;
  auto carve = [&](size_t bytes) {
    char* p = ws + off;
    off += (bytes + 255) & ~(size_t)255;
    return p;
  };
  const size_t NT = (size_t)Bc * Sc * DMc;  // 6.29M
  float* xf   = (float*)carve(NT * 4);
  bf16* xh    = (bf16*)carve(NT * 2);
  bf16* qh    = (bf16*)carve(NT * 2);
  bf16* kh    = (bf16*)carve(NT * 2);
  bf16* vh    = (bf16*)carve(NT * 2);
  bf16* kgh   = (bf16*)carve(NT * 2);
  bf16* vgh   = (bf16*)carve(NT * 2);
  bf16* vT    = (bf16*)carve((size_t)Bc * Hc * Dc * SPc * 2);
  bf16* ath   = (bf16*)carve(NT * 2);
  float* tmpf = (float*)carve(NT * 4);
  bf16* hh    = (bf16*)carve((size_t)Bc * Sc * DFFc * 2);
  bf16* wt    = (bf16*)carve((size_t)DFFc * DMc * 2);
  bf16* x0    = (bf16*)carve((size_t)16 * DMc * 2);
  float* qgb  = (float*)carve((size_t)16 * DMc * 4);

  const int rows = Bc * Sc;  // 8192
  dim3 blk(256);

  // embeddings + LN
  k_embed_ln<<<rows, blk, 0, stream>>>(ids, wemb, pemb, ln_es, ln_eb, xf, xh);

  auto transpose = [&](const float* w, int K, int N) {
    int total = K * N;
    k_transpose_w<<<(total + 255) / 256, blk, 0, stream>>>(w, wt, K, N);
  };
  auto gemm = [&](const bf16* A, const float* bias, float* oF, bf16* oH, int M,
                  int N, int K, float alpha, int gelu) {
    dim3 grid((M + 127) / 128, N / 128);
    k_gemm_bt<<<grid, blk, 0, stream>>>(A, wt, bias, oF, oH, M, N, K, alpha, gelu);
  };

  const float scale = 0.125f;  // 1/sqrt(64)

  for (int l = 0; l < Lc; l++) {
    const float* Wq_l = Wq + (size_t)l * DMc * DMc;
    const float* Wk_l = Wk + (size_t)l * DMc * DMc;
    const float* Wv_l = Wv + (size_t)l * DMc * DMc;
    const float* Wqg_l = Wqg + (size_t)l * DMc * DMc;
    const float* Wkg_l = Wkg + (size_t)l * DMc * DMc;
    const float* Wvg_l = Wvg + (size_t)l * DMc * DMc;
    const float* Wo_l = Wo + (size_t)l * DMc * DMc;
    const float* W1_l = W1 + (size_t)l * DMc * DFFc;
    const float* W2_l = W2 + (size_t)l * DFFc * DMc;

    // q/k/v projections (q pre-scaled)
    transpose(Wq_l, DMc, DMc);
    gemm(xh, bq + l * DMc, nullptr, qh, rows, DMc, DMc, scale, 0);
    transpose(Wk_l, DMc, DMc);
    gemm(xh, bk + l * DMc, nullptr, kh, rows, DMc, DMc, 1.0f, 0);
    transpose(Wv_l, DMc, DMc);
    gemm(xh, bv + l * DMc, nullptr, vh, rows, DMc, DMc, 1.0f, 0);

    // banded attention
    {
      int total = Bc * Hc * Dc * SPc;
      k_build_vT<<<(total + 255) / 256, blk, 0, stream>>>(vh, vT);
    }
    k_band_attn<<<Bc * Hc * NBc, blk, 0, stream>>>(qh, kh, vh, vT, am, ath);

    // global attention for token 0
    transpose(Wkg_l, DMc, DMc);
    gemm(xh, bkg + l * DMc, nullptr, kgh, rows, DMc, DMc, 1.0f, 0);
    transpose(Wvg_l, DMc, DMc);
    gemm(xh, bvg + l * DMc, nullptr, vgh, rows, DMc, DMc, 1.0f, 0);
    k_x0pad<<<(16 * DMc + 255) / 256, blk, 0, stream>>>(xh, x0);
    transpose(Wqg_l, DMc, DMc);
    gemm(x0, bqg + l * DMc, qgb, nullptr, 16, DMc, DMc, scale, 0);
    k_global_attn<<<Bc * Hc, blk, 0, stream>>>(qgb, kgh, vgh, am, ath);

    // output projection + residual LN
    transpose(Wo_l, DMc, DMc);
    gemm(ath, bo + l * DMc, tmpf, nullptr, rows, DMc, DMc, 1.0f, 0);
    k_add_ln<<<rows, blk, 0, stream>>>(xf, tmpf, l1s + l * DMc, l1b + l * DMc, xf, xh);

    // FFN
    transpose(W1_l, DMc, DFFc);
    gemm(xh, b1 + l * DFFc, nullptr, hh, rows, DFFc, DMc, 1.0f, 1);
    transpose(W2_l, DFFc, DMc);
    gemm(hh, b2 + l * DMc, tmpf, nullptr, rows, DMc, DFFc, 1.0f, 0);
    k_add_ln<<<rows, blk, 0, stream>>>(xf, tmpf, l2s + l * DMc, l2b + l * DMc, xf, xh);
  }

  // classifier head
  k_cls<<<1, blk, 0, stream>>>(xf, Wcls, bcls, out);
}
